// ImprovedGRUModel_32427003084945
// MI455X (gfx1250) — matrix operand order repaired and checked by reading, not compiled, not run on hardware
//
#include <hip/hip_runtime.h>
#include <hip/hip_bf16.h>

typedef _Float16 v16h __attribute__((ext_vector_type(16)));
typedef _Float16 v8h  __attribute__((ext_vector_type(8)));
typedef float    v8f  __attribute__((ext_vector_type(8)));
typedef unsigned int u32x4 __attribute__((ext_vector_type(4)));
typedef int          i32x4 __attribute__((ext_vector_type(4)));
typedef int          i32x8 __attribute__((ext_vector_type(8)));

#define DEV __device__ __forceinline__

static constexpr int B = 64, T = 1024, I = 64, H = 256, G = 768, O = 128;
static constexpr int M = B * T;   // 65536 rows for the batched GEMMs
static constexpr int NWG = 16;    // persistent workgroups in the recurrent scan

// ---------------------------------------------------------------- WMMA helpers

DEV v8f wmma16(v16h a, v16h b, v8f c) {
    // (neg_a, A, neg_b, B, c_mod, C, reuse_a, reuse_b)
    return __builtin_amdgcn_wmma_f32_16x16x32_f16(false, a, false, b, (short)0, c, false, false);
}

// A fragment: 16x32 f16 tile of row-major A (leading dim lda).
// Lane L<16 holds row M=L, K in {0..15}; lane L>=16 holds row M=L-16,
// K in {16..31}  -> two contiguous 8-half (16B) loads per lane.
DEV v16h load_a(const _Float16* __restrict__ A, int lda, int m0, int k0, int lane) {
    const _Float16* p = A + (size_t)(m0 + (lane & 15)) * lda + k0 + ((lane >> 4) << 4);
    v8h lo = *(const v8h*)p;         // K = base .. base+7
    v8h hi = *(const v8h*)(p + 8);   // K = base+8 .. base+15
    v16h a;
#pragma unroll
    for (int i = 0; i < 8; ++i) { a[i] = lo[i]; a[8 + i] = hi[i]; }
    return a;
}

// B fragment: 32x16 tile with B[k][n] = W[n0+n][k], W row-major [N,K].
// Lane = column N (low/high half select K rows 0..15 / 16..31): one 32B load.
DEV v16h load_b(const _Float16* __restrict__ W, int ldw, int n0, int k0, int lane) {
    const _Float16* p = W + (size_t)(n0 + (lane & 15)) * ldw + k0 + ((lane >> 4) << 4);
    return *(const v16h*)p;
}

DEV float fsigmoid(float x) { return 1.0f / (1.0f + __expf(-x)); }

// ------------------------------------------------------- Tensor Data Mover load
// 2D tile DMA: rows x cols f16 from global (row stride = stride elems) into LDS
// at byte offset lds_off. D# layout per CDNA5 ISA ch.8 (group0 + group1, 2D).
DEV void tdm_load_2d_f16(unsigned lds_off, const _Float16* gaddr,
                         int rows, int cols, int stride) {
    unsigned long long ga = (unsigned long long)(size_t)gaddr;
    u32x4 g0;
    g0[0] = 1u;                                         // count=1, user descriptor
    g0[1] = lds_off;                                    // lds_addr (bytes)
    g0[2] = (unsigned)(ga & 0xFFFFFFFFu);               // global_addr[31:0] -> bits 95:64
    g0[3] = (unsigned)((ga >> 32) & 0x01FFFFFFu)        // global_addr[56:32] -> bits 120:96
          | 0x80000000u;                                // type=2 (bits 127:126 = 10b)
    i32x8 g1;
    g1[0] = 0x00010000;                                 // data_size=1 (2 bytes), mask=0
    g1[1] = (cols & 0xFFFF) << 16;                      // tensor_dim0[15:0]
    g1[2] = ((cols >> 16) & 0xFFFF) | ((rows & 0xFFFF) << 16);  // dim0 hi | tensor_dim1 lo
    g1[3] = ((rows >> 16) & 0xFFFF) | ((cols & 0xFFFF) << 16);  // dim1 hi | tile_dim0
    g1[4] = rows & 0xFFFF;                              // tile_dim1 (tile_dim2 = 0)
    g1[5] = stride;                                     // tensor_dim0_stride[31:0]
    g1[6] = 0;                                          // stride hi | dim1_stride lo
    g1[7] = 0;
    i32x4 gz = {0, 0, 0, 0};                            // groups 2/3 unused (2D)
#if defined(__clang_major__) && __clang_major__ >= 23
    i32x8 gz8 = {0, 0, 0, 0, 0, 0, 0, 0};
    __builtin_amdgcn_tensor_load_to_lds(g0, g1, gz, gz, gz8, 0);
#else
    __builtin_amdgcn_tensor_load_to_lds(g0, g1, gz, gz, 0);
#endif
}

// ---------------------------------------------------------------- small utils

__global__ void zero_u32(unsigned* __restrict__ p, int n) {
    int i = blockIdx.x * blockDim.x + threadIdx.x;
    if (i < n) p[i] = 0u;
}

__global__ void f32_to_f16(const float* __restrict__ x, _Float16* __restrict__ y, int n) {
    int i = blockIdx.x * blockDim.x + threadIdx.x;
    if (i < n) y[i] = (_Float16)x[i];
}

// comb[g] = b_ih[g] + b_hh[g] for r,z gates (g < 2H); n-gate b_hh applied in scan
__global__ void prep_bias(const float* __restrict__ bih, const float* __restrict__ bhh,
                          float* __restrict__ comb) {
    int g = blockIdx.x * blockDim.x + threadIdx.x;
    if (g < G) comb[g] = bih[g] + (g < 2 * H ? bhh[g] : 0.0f);
}

// ---------------------------------------------------------------- batched GEMM
// C[M,N] = A_f16[M,K] * W_f16[N,K]^T + bias[N]; block = 4 waves, 64x64 C tile.
__global__ void __launch_bounds__(128) gemm_f16(
        const _Float16* __restrict__ A, const _Float16* __restrict__ W,
        const float* __restrict__ bias, float* __restrict__ C, int N, int K) {
    const int lane = threadIdx.x & 31;
    const int wave = threadIdx.x >> 5;
    const int nb = N >> 6;
    const int m0 = (blockIdx.x / nb) * 64 + wave * 16;
    const int n0 = (blockIdx.x % nb) * 64;

    v8f acc[4] = {};
    for (int k0 = 0; k0 < K; k0 += 32) {
        v16h a = load_a(A, K, m0, k0, lane);
        if (k0 + 32 < K)  // gfx1250 global_prefetch_b8 for the next A chunk
            __builtin_prefetch(A + (size_t)(m0 + (lane & 15)) * K + k0 + 32, 0, 1);
#pragma unroll
        for (int j = 0; j < 4; ++j) {
            v16h b = load_b(W, K, n0 + j * 16, k0, lane);
            acc[j] = wmma16(a, b, acc[j]);
        }
    }
    const int nl = lane & 15;
    const int mb = (lane >> 4) << 3;
#pragma unroll
    for (int j = 0; j < 4; ++j) {
        const float bv = bias[n0 + j * 16 + nl];
#pragma unroll
        for (int i = 0; i < 8; ++i)
            C[(size_t)(m0 + mb + i) * N + (n0 + j * 16 + nl)] = acc[j][i] + bv;
    }
}

// ---------------------------------------------------------------- device barrier

DEV void grid_barrier(unsigned* __restrict__ bar, unsigned* phase) {
    __syncthreads();
    // CDNA5 cluster barrier: real HW path under cluster dispatch, S_NOP otherwise.
    __builtin_amdgcn_s_cluster_barrier();
    if (threadIdx.x == 0) {
        const unsigned p = *phase;
        unsigned old = __hip_atomic_fetch_add(&bar[0], 1u, __ATOMIC_ACQ_REL,
                                              __HIP_MEMORY_SCOPE_AGENT);
        if (old == NWG - 1) {
            __hip_atomic_store(&bar[0], 0u, __ATOMIC_RELAXED, __HIP_MEMORY_SCOPE_AGENT);
            __hip_atomic_fetch_add(&bar[1], 1u, __ATOMIC_RELEASE, __HIP_MEMORY_SCOPE_AGENT);
        } else {
            while (__hip_atomic_load(&bar[1], __ATOMIC_ACQUIRE,
                                     __HIP_MEMORY_SCOPE_AGENT) == p)
                __builtin_amdgcn_s_sleep(8);
        }
    }
    *phase += 1;
    __syncthreads();
    __threadfence();  // acquire in every lane: invalidate WGP$ before reading h
}

// ---------------------------------------------------------------- recurrent scan
// Persistent: 16 WGs x 128 threads. WG owns 16 hidden units (j0), wave owns 16
// batch rows (m0). W_hh rows DMA'd into LDS by the Tensor Data Mover; h state
// double-buffered f16 in global (L2-resident); h_prev f32 kept in registers.
__global__ void __launch_bounds__(128) gru_scan(
        const float* __restrict__ xw,      // [B,T,3H], b_ih + b_hh(r,z) folded in
        const _Float16* __restrict__ whh,  // [3H,H] f16
        const float* __restrict__ bhh,     // [3H] (n-gate part used here)
        float* __restrict__ hout,          // [B,T,H] f32
        _Float16* __restrict__ hbuf,       // [2][B*H] f16 double buffer
        unsigned* __restrict__ bar, int steps) {
    __shared__ _Float16 sB[3 * 16 * H];  // 24 KB: rows (gate,j) x K, row-major
    const int lane = threadIdx.x & 31;
    const int wave = threadIdx.x >> 5;
    const int wg = blockIdx.x;
    const int j0 = wg * 16;
    const int m0 = wave * 16;
    const int nl = lane & 15;
    const int mb = (lane >> 4) << 3;

    if (wave == 0) {
        // TDM: three 16x256 f16 tiles (one per gate), row stride = H
        const unsigned lds0 = (unsigned)(size_t)&sB[0];
#pragma unroll
        for (int g = 0; g < 3; ++g)
            tdm_load_2d_f16(lds0 + (unsigned)(g * 16 * H * 2),
                            whh + (size_t)(g * H + j0) * H, 16, H, H);
        __builtin_amdgcn_s_wait_tensorcnt(0);
    }
    for (int i = wg * 128 + threadIdx.x; i < 2 * B * H; i += NWG * 128)
        hbuf[i] = (_Float16)0.0f;  // h(-1) = 0 in both buffers
    __threadfence();
    unsigned phase = 0;
    grid_barrier(bar, &phase);

    const float bnn = bhh[2 * H + j0 + nl];
    float hprev[8];
#pragma unroll
    for (int i = 0; i < 8; ++i) hprev[i] = 0.0f;

    const int kh = (lane >> 4) << 4;  // K sub-offset for B fragments

    for (int t = 0; t < steps; ++t) {
        const _Float16* hc = hbuf + (size_t)(t & 1) * (B * H);
        _Float16* hn = hbuf + (size_t)((t + 1) & 1) * (B * H);

        v8f ar = {}, az = {}, an = {};
#pragma unroll
        for (int kc = 0; kc < 8; ++kc) {
            v16h a = load_a(hc, H, m0, kc * 32, lane);
            v16h b0 = *(const v16h*)&sB[(0 * 16 + nl) * H + kc * 32 + kh];
            v16h b1 = *(const v16h*)&sB[(1 * 16 + nl) * H + kc * 32 + kh];
            v16h b2 = *(const v16h*)&sB[(2 * 16 + nl) * H + kc * 32 + kh];
            ar = wmma16(a, b0, ar);
            az = wmma16(a, b1, az);
            an = wmma16(a, b2, an);
        }
#pragma unroll
        for (int i = 0; i < 8; ++i) {
            const int b = m0 + mb + i;
            const float* xrow = xw + ((size_t)b * T + t) * G;
            if (t + 1 < steps) {  // prefetch next timestep's gate inputs
                const float* xn = xrow + G + j0 + nl;
                __builtin_prefetch(xn, 0, 1);
                __builtin_prefetch(xn + H, 0, 1);
                __builtin_prefetch(xn + 2 * H, 0, 1);
            }
            const float r = fsigmoid(xrow[j0 + nl] + ar[i]);
            const float z = fsigmoid(xrow[H + j0 + nl] + az[i]);
            const float n = tanhf(xrow[2 * H + j0 + nl] + r * (an[i] + bnn));
            const float h = (1.0f - z) * n + z * hprev[i];
            hprev[i] = h;
            hout[((size_t)b * T + t) * H + j0 + nl] = h;
            hn[(size_t)b * H + j0 + nl] = (_Float16)h;
        }
        __threadfence();
        grid_barrier(bar, &phase);
    }
}

// ---------------------------------------------------------------- BN / pool / FC

// grid = M/256 blocks of 256 threads; thread t owns channel t for 256 rows
__global__ void bn_stats(const float* __restrict__ x, float* __restrict__ sum,
                         float* __restrict__ sq) {
    const int k = threadIdx.x;
    const size_t r0 = (size_t)blockIdx.x * 256;
    float s = 0.0f, q = 0.0f;
    for (int r = 0; r < 256; ++r) {
        float v = x[(r0 + r) * H + k];
        s += v;
        q += v * v;
    }
    atomicAdd(&sum[k], s);
    atomicAdd(&sq[k], q);
}

__global__ void bn_finalize(const float* __restrict__ sum, const float* __restrict__ sq,
                            const float* __restrict__ gamma, const float* __restrict__ beta,
                            float* __restrict__ s, float* __restrict__ t, int rows) {
    const int k = threadIdx.x;
    const float inv = 1.0f / (float)rows;
    const float mu = sum[k] * inv;
    const float var = sq[k] * inv - mu * mu;
    const float sc = gamma[k] * rsqrtf(var + 1e-5f);
    s[k] = sc;
    t[k] = beta[k] - mu * sc;
}

// y_f16[i] = f16(x[i]*s[ch] + t[ch])   (BN fused into layer-2 A-matrix conversion)
__global__ void bn_apply_f16(const float* __restrict__ x, const float* __restrict__ s,
                             const float* __restrict__ t, _Float16* __restrict__ y) {
    size_t i = (size_t)blockIdx.x * blockDim.x + threadIdx.x;
    if (i < (size_t)M * H) {
        int k = (int)(i & (H - 1));
        y[i] = (_Float16)(x[i] * s[k] + t[k]);
    }
}

// p[b][k] = tanh(max_t (h2*s+t));  grid=B blocks x H threads (coalesced over k)
__global__ void maxpool_tanh(const float* __restrict__ h2, const float* __restrict__ s,
                             const float* __restrict__ t, float* __restrict__ p) {
    const int b = blockIdx.x, k = threadIdx.x;
    const float sc = s[k], tt = t[k];
    const float* base = h2 + (size_t)b * T * H + k;
    float m = -3.402823466e+38f;
    for (int i = 0; i < T; ++i) m = fmaxf(m, base[(size_t)i * H] * sc + tt);
    p[b * H + k] = tanhf(m);
}

// out[b][o] = p[b,:] . w_fc[o,:] + b_fc[o]   (tiny: 4 MFLOP)
__global__ void fc_out(const float* __restrict__ p, const float* __restrict__ wfc,
                       const float* __restrict__ bfc, float* __restrict__ out) {
    const int b = blockIdx.x, o = threadIdx.x;
    const float* pr = p + b * H;
    const float* wr = wfc + o * H;
    float acc = bfc[o];
    for (int k = 0; k < H; ++k) acc += pr[k] * wr[k];
    out[b * O + o] = acc;
}

// ---------------------------------------------------------------- host driver

extern "C" void kernel_launch(void* const* d_in, const int* in_sizes, int n_in,
                              void* d_out, int out_size, void* d_ws, size_t ws_size,
                              hipStream_t stream) {
    (void)in_sizes; (void)n_in; (void)out_size; (void)ws_size;
    const float* x     = (const float*)d_in[0];
    const float* w_ih1 = (const float*)d_in[1];
    const float* w_hh1 = (const float*)d_in[2];
    const float* b_ih1 = (const float*)d_in[3];
    const float* b_hh1 = (const float*)d_in[4];
    const float* g1    = (const float*)d_in[5];
    const float* be1   = (const float*)d_in[6];
    const float* w_ih2 = (const float*)d_in[7];
    const float* w_hh2 = (const float*)d_in[8];
    const float* b_ih2 = (const float*)d_in[9];
    const float* b_hh2 = (const float*)d_in[10];
    const float* g2    = (const float*)d_in[11];
    const float* be2   = (const float*)d_in[12];
    const float* w_fc  = (const float*)d_in[13];
    const float* b_fc  = (const float*)d_in[14];
    float* out = (float*)d_out;

    char* cur = (char*)d_ws;
    auto alloc = [&](size_t bytes) -> char* {
        char* p = cur;
        cur += (bytes + 255) & ~(size_t)255;
        return p;
    };
    // small region (zeroed every call): barriers + BN accumulators
    unsigned* bar1 = (unsigned*)alloc(64);
    unsigned* bar2 = (unsigned*)alloc(64);
    float* sum1 = (float*)alloc(H * 4);
    float* sq1  = (float*)alloc(H * 4);
    float* sum2 = (float*)alloc(H * 4);
    float* sq2  = (float*)alloc(H * 4);
    // derived params (fully written each call)
    float* s1 = (float*)alloc(H * 4);
    float* t1 = (float*)alloc(H * 4);
    float* s2 = (float*)alloc(H * 4);
    float* t2 = (float*)alloc(H * 4);
    float* comb1 = (float*)alloc(G * 4);
    float* comb2 = (float*)alloc(G * 4);
    float* pool = (float*)alloc((size_t)B * H * 4);
    // big buffers
    _Float16* xf16  = (_Float16*)alloc((size_t)M * I * 2);
    _Float16* wi1h  = (_Float16*)alloc((size_t)G * I * 2);
    _Float16* wh1h  = (_Float16*)alloc((size_t)G * H * 2);
    _Float16* wi2h  = (_Float16*)alloc((size_t)G * H * 2);
    _Float16* wh2h  = (_Float16*)alloc((size_t)G * H * 2);
    _Float16* hbuf  = (_Float16*)alloc((size_t)2 * B * H * 2);
    float*    xw    = (float*)alloc((size_t)M * G * 4);  // reused for both layers
    float*    h1    = (float*)alloc((size_t)M * H * 4);
    _Float16* a2f16 = (_Float16*)alloc((size_t)M * H * 2);
    float*    h2    = (float*)alloc((size_t)M * H * 4);

    // 0) zero barriers + BN accumulators (first 8 KB of ws)
    zero_u32<<<8, 256, 0, stream>>>((unsigned*)d_ws, 2048);

    // 1) f16 conversions of activations/weights
    f32_to_f16<<<(M * I + 255) / 256, 256, 0, stream>>>(x, xf16, M * I);
    f32_to_f16<<<(G * I + 255) / 256, 256, 0, stream>>>(w_ih1, wi1h, G * I);
    f32_to_f16<<<(G * H + 255) / 256, 256, 0, stream>>>(w_hh1, wh1h, G * H);
    f32_to_f16<<<(G * H + 255) / 256, 256, 0, stream>>>(w_ih2, wi2h, G * H);
    f32_to_f16<<<(G * H + 255) / 256, 256, 0, stream>>>(w_hh2, wh2h, G * H);
    prep_bias<<<(G + 255) / 256, 256, 0, stream>>>(b_ih1, b_hh1, comb1);
    prep_bias<<<(G + 255) / 256, 256, 0, stream>>>(b_ih2, b_hh2, comb2);

    const int gemm_grid = (M / 64) * (G / 64);  // 12288 blocks

    // 2) layer-1 input GEMM: xw = x @ w_ih1^T + comb1   [M, 3H], K = I
    gemm_f16<<<gemm_grid, 128, 0, stream>>>(xf16, wi1h, comb1, xw, G, I);
    // 3) layer-1 recurrent scan (persistent, 16 WGs)
    gru_scan<<<NWG, 128, 0, stream>>>(xw, wh1h, b_hh1, h1, hbuf, bar1, T);

    // 4) BN1 stats + fused BN->f16 conversion of layer-2 input
    bn_stats<<<M / 256, 256, 0, stream>>>(h1, sum1, sq1);
    bn_finalize<<<1, 256, 0, stream>>>(sum1, sq1, g1, be1, s1, t1, M);
    bn_apply_f16<<<(int)(((size_t)M * H + 255) / 256), 256, 0, stream>>>(h1, s1, t1, a2f16);

    // 5) layer-2 input GEMM: xw = bn(h1) @ w_ih2^T + comb2   [M, 3H], K = H
    gemm_f16<<<gemm_grid, 128, 0, stream>>>(a2f16, wi2h, comb2, xw, G, H);
    // 6) layer-2 recurrent scan
    gru_scan<<<NWG, 128, 0, stream>>>(xw, wh2h, b_hh2, h2, hbuf, bar2, T);

    // 7) BN2 stats + max-over-time + tanh + FC
    bn_stats<<<M / 256, 256, 0, stream>>>(h2, sum2, sq2);
    bn_finalize<<<1, 256, 0, stream>>>(sum2, sq2, g2, be2, s2, t2, M);
    maxpool_tanh<<<B, H, 0, stream>>>(h2, s2, t2, pool);
    fc_out<<<B, O, 0, stream>>>(pool, w_fc, b_fc, out);
}
